// TransformerBlock_7954279432840
// MI455X (gfx1250) — compile-verified, matrix operand order checked by reading, not run on hardware
//
#include <hip/hip_runtime.h>
#include <math.h>

// ---------------------------------------------------------------------------
// Types for CDNA5 WMMA (wave32, 16x16x32 bf16 -> f32)
// ---------------------------------------------------------------------------
typedef __attribute__((ext_vector_type(16))) __bf16 v16bf;
typedef __attribute__((ext_vector_type(8)))  float  v8f;
typedef __attribute__((ext_vector_type(4)))  int    v4i;

__device__ __forceinline__ v8f wmma_bf16(v16bf a, v16bf b, v8f c) {
  // (neg_a, A, neg_b, B, c_mod, C, reuse_a, reuse_b)
  return __builtin_amdgcn_wmma_f32_16x16x32_bf16(false, a, false, b,
                                                 (short)0, c, false, false);
}

// CDNA5 async global->LDS copies (ASYNCcnt).  Signature confirmed by the
// round-2 diagnostic: param0 = addrspace(1) int4*, param1 = lds int4*.
// Used unconditionally on gfx1250 so the compile result verifies emission.
#if defined(__gfx1250__)
#define HAVE_ASYNC 1
typedef __attribute__((address_space(1))) v4i gv4i_t;   // global int4
typedef __attribute__((address_space(3))) v4i lv4i_t;   // LDS int4
#else
#define HAVE_ASYNC 0
#endif

#define EPSV 1e-5f

// Load a 32-byte bf16 fragment from LDS via two 16B chunks (16B alignment).
__device__ __forceinline__ v16bf load_frag16(const __bf16* p) {
  union { uint4 u[2]; v16bf v; } t;
  t.u[0] = *(const uint4*)p;
  t.u[1] = *(const uint4*)(p + 8);
  return t.v;
}

// ---------------------------------------------------------------------------
// fp32 -> bf16 elementwise conversion (weights / misc)
// ---------------------------------------------------------------------------
__global__ void f32_to_bf16_kernel(const float* __restrict__ in,
                                   __bf16* __restrict__ out, size_t n) {
  size_t i = (size_t)blockIdx.x * blockDim.x + threadIdx.x;
  if (i < n) out[i] = (__bf16)in[i];
}

// ---------------------------------------------------------------------------
// LayerNorm: fp32 [rows, C] -> bf16 [rows, C].  One block (256 thr) per row.
// ---------------------------------------------------------------------------
__global__ void ln_bf16_kernel(const float* __restrict__ x,
                               const float* __restrict__ w,
                               const float* __restrict__ b,
                               __bf16* __restrict__ out, int C) {
  int row = blockIdx.x;
  int tid = threadIdx.x;
  const float* xr = x + (size_t)row * C;
  __shared__ float rs[256];
  __shared__ float rs2[256];
  float s = 0.f, s2 = 0.f;
  for (int i = tid; i < C; i += blockDim.x) {
    float v = xr[i];
    s += v; s2 += v * v;
  }
  rs[tid] = s; rs2[tid] = s2;
  __syncthreads();
  for (int off = 128; off > 0; off >>= 1) {
    if (tid < off) { rs[tid] += rs[tid + off]; rs2[tid] += rs2[tid + off]; }
    __syncthreads();
  }
  float mu  = rs[0] / (float)C;
  float var = rs2[0] / (float)C - mu * mu;
  float rstd = rsqrtf(var + EPSV);
  __bf16* orow = out + (size_t)row * C;
  for (int i = tid; i < C; i += blockDim.x) {
    orow[i] = (__bf16)((xr[i] - mu) * rstd * w[i] + b[i]);
  }
}

// ---------------------------------------------------------------------------
// WMMA GEMM:  out[M,N] = A[M,K] @ W[N,K]^T   (both bf16, row-major along K)
//   block = 256 threads = 8 waves (4 x 2), block tile 128(M) x 64(N), K-step 32.
//   A 128x32 and W 64x32 K-slices are staged into LDS with double buffering;
//   staging uses GLOBAL_LOAD_ASYNC_TO_LDS_B128 (3 per thread per stage)
//   retired with s_wait_asynccnt(3) so stage k+1 overlaps compute of k.
//   LDS rows padded to 40 bf16 (80B) to spread ds_load_b128 across banks.
// Fragment addressing (wave32):
//   A frag : lane L -> row (L&15), 16 halves at k-slice + (L>>4)*16
//   B frag : lane L -> row (L&15) of W tile, same K slice
//   C/D    : lane L, vgpr r -> (M = r + 8*(L>>4), N = L&15)
// ---------------------------------------------------------------------------
#define LDSK 40   // padded row stride (bf16 elements) for 32-wide K slice

template <bool OUT_BF16>
__global__ void gemm_bf16_wmma_kernel(const __bf16* __restrict__ A,
                                      const __bf16* __restrict__ W,
                                      void* __restrict__ outp,
                                      const float* __restrict__ res,
                                      int M, int N, int K) {
  const int tid  = threadIdx.x;
  const int lane = tid & 31;
  const int wave = tid >> 5;              // 0..7
  const int wm = wave & 3;                // 0..3
  const int wn = wave >> 2;               // 0..1
  const int mblk = blockIdx.y * 128;
  const int nblk = blockIdx.x * 64;
  const int hi = lane >> 4;               // 0/1
  const int ln = lane & 15;

  __shared__ __attribute__((aligned(32))) __bf16 As[2][128 * LDSK];
  __shared__ __attribute__((aligned(32))) __bf16 Bs[2][64 * LDSK];

  // cooperative staging of one K-slice (A: 512 b128 vectors, B: 256)
  auto stage = [&](int buf, int k0) {
#pragma unroll
    for (int i = 0; i < 2; ++i) {
      int v = tid + 256 * i;
      int r = v >> 2, c = (v & 3) * 8;
      const __bf16* gp = A + (size_t)(mblk + r) * K + k0 + c;
      __bf16* lp = &As[buf][r * LDSK + c];
#if HAVE_ASYNC
      __builtin_amdgcn_global_load_async_to_lds_b128((gv4i_t*)gp, (lv4i_t*)lp, 0, 0);
#else
      *(uint4*)lp = *(const uint4*)gp;
#endif
    }
    {
      int r = tid >> 2, c = (tid & 3) * 8;
      const __bf16* gp = W + (size_t)(nblk + r) * K + k0 + c;
      __bf16* lp = &Bs[buf][r * LDSK + c];
#if HAVE_ASYNC
      __builtin_amdgcn_global_load_async_to_lds_b128((gv4i_t*)gp, (lv4i_t*)lp, 0, 0);
#else
      *(uint4*)lp = *(const uint4*)gp;
#endif
    }
  };

  v8f acc00 = {}, acc01 = {}, acc10 = {}, acc11 = {};

  stage(0, 0);
  int buf = 0;
  for (int k0 = 0; k0 < K; k0 += 32) {
    const bool has_next = (k0 + 32 < K);
    if (has_next) stage(buf ^ 1, k0 + 32);
#if HAVE_ASYNC
    if (has_next) __builtin_amdgcn_s_wait_asynccnt(3);  // retire current buf
    else          __builtin_amdgcn_s_wait_asynccnt(0);
#endif
    __syncthreads();   // current buf visible to all waves

    const __bf16* Ab = &As[buf][0];
    const __bf16* Bb = &Bs[buf][0];
    v16bf a0 = load_frag16(Ab + (wm * 32 + ln)      * LDSK + hi * 16);
    v16bf a1 = load_frag16(Ab + (wm * 32 + 16 + ln) * LDSK + hi * 16);
    v16bf b0 = load_frag16(Bb + (wn * 32 + ln)      * LDSK + hi * 16);
    v16bf b1 = load_frag16(Bb + (wn * 32 + 16 + ln) * LDSK + hi * 16);
    acc00 = wmma_bf16(a0, b0, acc00);
    acc01 = wmma_bf16(a0, b1, acc01);
    acc10 = wmma_bf16(a1, b0, acc10);
    acc11 = wmma_bf16(a1, b1, acc11);

    __syncthreads();   // all reads done before buf is restaged next iter
    buf ^= 1;
  }

  const int m0 = mblk + wm * 32;
  const int n0 = nblk + wn * 32;
  for (int tm = 0; tm < 2; ++tm) {
    for (int tn = 0; tn < 2; ++tn) {
      v8f acc = (tm == 0) ? (tn == 0 ? acc00 : acc01)
                          : (tn == 0 ? acc10 : acc11);
      for (int r = 0; r < 8; ++r) {
        int row = m0 + tm * 16 + r + 8 * hi;
        int col = n0 + tn * 16 + ln;
        size_t idx = (size_t)row * N + col;
        float v = acc[r];
        if (res) v += res[idx];
        if (OUT_BF16) ((__bf16*)outp)[idx] = (__bf16)v;
        else          ((float*)outp)[idx]  = v;
      }
    }
  }
}

// ---------------------------------------------------------------------------
// Split fused QKV (bf16 [B*T, 3C]) into:
//   Q  [B,H,T,Dh] (bf16, pre-scaled by 1/sqrt(Dh))
//   K  [B,H,T,Dh] (bf16)
//   Vt [B,H,Dh,T] (bf16, transposed so PV B-fragments load contiguously)
// ---------------------------------------------------------------------------
__global__ void qkv_split_kernel(const __bf16* __restrict__ qkv,
                                 __bf16* __restrict__ Q,
                                 __bf16* __restrict__ Kd,
                                 __bf16* __restrict__ Vt,
                                 int B, int T, int H, int Dh) {
  size_t total = (size_t)B * T * H * Dh;
  size_t idx = (size_t)blockIdx.x * blockDim.x + threadIdx.x;
  if (idx >= total) return;
  int d = (int)(idx % Dh);
  size_t t1 = idx / Dh;
  int h = (int)(t1 % H);
  size_t t2 = t1 / H;
  int t = (int)(t2 % T);
  int b = (int)(t2 / T);
  int C = H * Dh;
  size_t base = ((size_t)b * T + t) * (size_t)(3 * C) + (size_t)h * Dh + d;
  float q = (float)qkv[base] * 0.125f;   // 1/sqrt(64)
  float k = (float)qkv[base + C];
  float v = (float)qkv[base + 2 * C];
  size_t o = (((size_t)(b * H + h) * T + t) * Dh + d);
  Q[o]  = (__bf16)q;
  Kd[o] = (__bf16)k;
  Vt[((size_t)(b * H + h) * Dh + d) * T + t] = (__bf16)v;
}

// ---------------------------------------------------------------------------
// Flash-style causal attention, one wave (32 thr) per 16-query tile.
//   Q  [B,H,T,64], K [B,H,T,64], Vt [B,H,64,T]  (all bf16)
//   O  [B*T, C] bf16 (head-interleaved, ready for the out-projection GEMM)
// Per 32-key step: 4 WMMAs for S = Q K^T, online softmax via LDS (lanes 0..15
// each own one query row), 4 WMMAs for acc += P V.  f32 accumulation.
// ---------------------------------------------------------------------------
__global__ void flash_attn_kernel(const __bf16* __restrict__ Q,
                                  const __bf16* __restrict__ Kd,
                                  const __bf16* __restrict__ Vt,
                                  __bf16* __restrict__ O,
                                  int T, int H) {
  const int Dh = 64;
  const int bh = blockIdx.x;          // b*H + h
  const int qt = blockIdx.y;          // query tile
  const int lane = threadIdx.x;       // 0..31
  const int hi = lane >> 4, ln = lane & 15;
  const int q0 = qt * 16;
  const int b = bh / H, h = bh % H;
  const int C = H * Dh;

  const __bf16* Qb = Q  + (size_t)bh * T * Dh;
  const __bf16* Kb = Kd + (size_t)bh * T * Dh;
  const __bf16* Vb = Vt + (size_t)bh * Dh * T;

  __shared__ float  Sbuf[16 * 32];
  __shared__ __attribute__((aligned(32))) __bf16 Pbuf[16 * 32];
  __shared__ float mrow[16], lrow[16], arow[16];

  // Q fragments (loaded once): 16x64 = two 16x32 A-fragments
  v16bf qa0 = *(const v16bf*)(Qb + (size_t)(q0 + ln) * Dh + 0  + hi * 16);
  v16bf qa1 = *(const v16bf*)(Qb + (size_t)(q0 + ln) * Dh + 32 + hi * 16);

  v8f acc0 = {}, acc1 = {}, acc2 = {}, acc3 = {};

  if (lane < 16) { mrow[lane] = -__builtin_inff(); lrow[lane] = 0.f; }
  __syncthreads();

  for (int kb = 0; kb <= q0 + 15; kb += 32) {
    // ---- S = Q @ K^T for keys [kb, kb+32): load all 4 K frags, then MMA ----
    v16bf k00 = *(const v16bf*)(Kb + (size_t)(kb + ln)      * Dh + 0  + hi * 16);
    v16bf k10 = *(const v16bf*)(Kb + (size_t)(kb + 16 + ln) * Dh + 0  + hi * 16);
    v16bf k01 = *(const v16bf*)(Kb + (size_t)(kb + ln)      * Dh + 32 + hi * 16);
    v16bf k11 = *(const v16bf*)(Kb + (size_t)(kb + 16 + ln) * Dh + 32 + hi * 16);
    v8f s0 = {}, s1 = {};
    s0 = wmma_bf16(qa0, k00, s0);
    s1 = wmma_bf16(qa0, k10, s1);
    s0 = wmma_bf16(qa1, k01, s0);
    s1 = wmma_bf16(qa1, k11, s1);

    // dump S fragment to LDS: frag row = r + 8*hi, col = ln (+16 for s1)
    for (int r = 0; r < 8; ++r) {
      Sbuf[(r + 8 * hi) * 32 + ln]      = s0[r];
      Sbuf[(r + 8 * hi) * 32 + 16 + ln] = s1[r];
    }
    __syncthreads();

    // ---- online softmax: lanes 0..15 each own one query row ----
    if (lane < 16) {
      int row = lane;
      int abs_row = q0 + row;
      float m_old = mrow[row];
      float mx = m_old;
      for (int c = 0; c < 32; ++c) {
        if (kb + c <= abs_row) mx = fmaxf(mx, Sbuf[row * 32 + c]);
      }
      float al = __expf(m_old - mx);          // m_old=-inf first iter -> 0
      float sum = 0.f;
      for (int c = 0; c < 32; ++c) {
        float p = (kb + c <= abs_row) ? __expf(Sbuf[row * 32 + c] - mx) : 0.f;
        sum += p;
        Pbuf[row * 32 + c] = (__bf16)p;
      }
      lrow[row] = lrow[row] * al + sum;
      mrow[row] = mx;
      arow[row] = al;
    }
    __syncthreads();

    // ---- rescale accumulators by alpha(row) ----
    for (int r = 0; r < 8; ++r) {
      float al = arow[r + 8 * hi];
      acc0[r] *= al; acc1[r] *= al; acc2[r] *= al; acc3[r] *= al;
    }

    // ---- acc += P (16x32) @ V (32x64) ----
    v16bf pa = *(const v16bf*)(Pbuf + ln * 32 + hi * 16);
    v16bf vb0 = *(const v16bf*)(Vb + (size_t)(0  + ln) * T + kb + hi * 16);
    v16bf vb1 = *(const v16bf*)(Vb + (size_t)(16 + ln) * T + kb + hi * 16);
    v16bf vb2 = *(const v16bf*)(Vb + (size_t)(32 + ln) * T + kb + hi * 16);
    v16bf vb3 = *(const v16bf*)(Vb + (size_t)(48 + ln) * T + kb + hi * 16);
    acc0 = wmma_bf16(pa, vb0, acc0);
    acc1 = wmma_bf16(pa, vb1, acc1);
    acc2 = wmma_bf16(pa, vb2, acc2);
    acc3 = wmma_bf16(pa, vb3, acc3);

    __syncthreads();   // protect Sbuf/Pbuf/arow before next iteration
  }

  // ---- epilogue: normalize by l, store bf16 [B*T, C] head-interleaved ----
  for (int r = 0; r < 8; ++r) {
    int row = r + 8 * hi;
    float inv_l = 1.f / lrow[row];
    size_t orow = ((size_t)b * T + q0 + row) * C + (size_t)h * Dh;
    O[orow + 0  + ln] = (__bf16)(acc0[r] * inv_l);
    O[orow + 16 + ln] = (__bf16)(acc1[r] * inv_l);
    O[orow + 32 + ln] = (__bf16)(acc2[r] * inv_l);
    O[orow + 48 + ln] = (__bf16)(acc3[r] * inv_l);
  }
}

// ---------------------------------------------------------------------------
// SwiGLU combine: a <- silu(a) * g   (bf16 in, bf16 out, in-place into a)
// ---------------------------------------------------------------------------
__global__ void swiglu_kernel(__bf16* __restrict__ a,
                              const __bf16* __restrict__ g, size_t n) {
  size_t i = (size_t)blockIdx.x * blockDim.x + threadIdx.x;
  if (i >= n) return;
  float x = (float)a[i];
  float y = (float)g[i];
  float s = x / (1.f + __expf(-x));
  a[i] = (__bf16)(s * y);
}

// ---------------------------------------------------------------------------
// Host launcher
// ---------------------------------------------------------------------------
extern "C" void kernel_launch(void* const* d_in, const int* in_sizes, int n_in,
                              void* d_out, int out_size, void* d_ws, size_t ws_size,
                              hipStream_t stream) {
  (void)in_sizes; (void)n_in; (void)out_size; (void)ws_size;

  const int B = 2, T = 2048, C = 1024, H = 16, Dh = 64, FF = 4096;
  const int M = B * T;                 // 4096 rows

  const float* x          = (const float*)d_in[0];
  const float* ln1_w      = (const float*)d_in[1];
  const float* ln1_b      = (const float*)d_in[2];
  const float* qkv_w      = (const float*)d_in[3];
  const float* attn_out_w = (const float*)d_in[4];
  const float* ln2_w      = (const float*)d_in[5];
  const float* ln2_b      = (const float*)d_in[6];
  const float* mlp_in_w   = (const float*)d_in[7];
  const float* mlp_gate_w = (const float*)d_in[8];
  const float* mlp_out_w  = (const float*)d_in[9];
  float* out = (float*)d_out;

  // ---- carve workspace (256B aligned slices) ----
  char* ws = (char*)d_ws;
  size_t off = 0;
  auto carve = [&](size_t bytes) -> char* {
    char* p = ws + off;
    off += (bytes + 255) & ~(size_t)255;
    return p;
  };
  __bf16* wqkv_bf  = (__bf16*)carve((size_t)3 * C * C * 2);
  __bf16* wout_bf  = (__bf16*)carve((size_t)C * C * 2);
  __bf16* win_bf   = (__bf16*)carve((size_t)FF * C * 2);
  __bf16* wgate_bf = (__bf16*)carve((size_t)FF * C * 2);
  __bf16* wo2_bf   = (__bf16*)carve((size_t)C * FF * 2);
  __bf16* hbf      = (__bf16*)carve((size_t)M * C * 2);
  __bf16* qkvbf    = (__bf16*)carve((size_t)M * 3 * C * 2);
  __bf16* Qbf      = (__bf16*)carve((size_t)M * C * 2);
  __bf16* Kbf      = (__bf16*)carve((size_t)M * C * 2);
  __bf16* Vtb      = (__bf16*)carve((size_t)M * C * 2);
  __bf16* attnbf   = (__bf16*)carve((size_t)M * C * 2);
  float*  x1       = (float*) carve((size_t)M * C * 4);
  __bf16* h2bf     = (__bf16*)carve((size_t)M * C * 2);
  __bf16* abf      = (__bf16*)carve((size_t)M * FF * 2);
  __bf16* gbf      = (__bf16*)carve((size_t)M * FF * 2);

  auto cdiv = [](size_t a, size_t b) { return (a + b - 1) / b; };

  // ---- 1) weights fp32 -> bf16 ----
  {
    struct { const float* src; __bf16* dst; size_t n; } cv[5] = {
      { qkv_w,      wqkv_bf,  (size_t)3 * C * C },
      { attn_out_w, wout_bf,  (size_t)C * C     },
      { mlp_in_w,   win_bf,   (size_t)FF * C    },
      { mlp_gate_w, wgate_bf, (size_t)FF * C    },
      { mlp_out_w,  wo2_bf,   (size_t)C * FF    },
    };
    for (int i = 0; i < 5; ++i)
      f32_to_bf16_kernel<<<cdiv(cv[i].n, 256), 256, 0, stream>>>(cv[i].src, cv[i].dst, cv[i].n);
  }

  // ---- 2) LN1 -> bf16 ----
  ln_bf16_kernel<<<M, 256, 0, stream>>>(x, ln1_w, ln1_b, hbf, C);

  // ---- 3) QKV GEMM: [M,C] @ [3C,C]^T -> bf16 [M,3C] ----
  gemm_bf16_wmma_kernel<true><<<dim3(3 * C / 64, M / 128), 256, 0, stream>>>(
      hbf, wqkv_bf, (void*)qkvbf, nullptr, M, 3 * C, C);

  // ---- 4) split QKV, scale Q, transpose V ----
  qkv_split_kernel<<<cdiv((size_t)M * C, 256), 256, 0, stream>>>(
      qkvbf, Qbf, Kbf, Vtb, B, T, H, Dh);

  // ---- 5) causal flash attention ----
  flash_attn_kernel<<<dim3(B * H, T / 16), 32, 0, stream>>>(Qbf, Kbf, Vtb, attnbf, T, H);

  // ---- 6) out-projection + residual: x1 = x + attn @ Wo^T ----
  gemm_bf16_wmma_kernel<false><<<dim3(C / 64, M / 128), 256, 0, stream>>>(
      attnbf, wout_bf, (void*)x1, x, M, C, C);

  // ---- 7) LN2 -> bf16 ----
  ln_bf16_kernel<<<M, 256, 0, stream>>>(x1, ln2_w, ln2_b, h2bf, C);

  // ---- 8/9) MLP in + gate GEMMs -> bf16 [M,FF] ----
  gemm_bf16_wmma_kernel<true><<<dim3(FF / 64, M / 128), 256, 0, stream>>>(
      h2bf, win_bf, (void*)abf, nullptr, M, FF, C);
  gemm_bf16_wmma_kernel<true><<<dim3(FF / 64, M / 128), 256, 0, stream>>>(
      h2bf, wgate_bf, (void*)gbf, nullptr, M, FF, C);

  // ---- 10) SwiGLU combine (in place into abf) ----
  swiglu_kernel<<<cdiv((size_t)M * FF, 256), 256, 0, stream>>>(abf, gbf, (size_t)M * FF);

  // ---- 11) MLP out GEMM + residual -> d_out (fp32) ----
  gemm_bf16_wmma_kernel<false><<<dim3(C / 64, M / 128), 256, 0, stream>>>(
      abf, wo2_bf, (void*)out, x1, M, C, FF);
}
